// InfiniteTransformerECG_89945205113181
// MI455X (gfx1250) — matrix operand order repaired and checked by reading, not compiled, not run on hardware
//
#include <hip/hip_runtime.h>
#include <hip/hip_bf16.h>
#include <math.h>

// ---------------------------------------------------------------------------
// MI455X (gfx1250, wave32) implementation of the Infini+Memorizing transformer
// forward pass. All dense math runs through v_wmma_f32_16x16x32_bf16 with f32
// accumulation; softmax/LN/normalize in f32. Two fusion kernels avoid the two
// 256MB intermediates (sims matrix, memorizing attention scores).
// ---------------------------------------------------------------------------

typedef __bf16 bf16;
typedef __attribute__((ext_vector_type(16))) __bf16 v16bf_t;
typedef __attribute__((ext_vector_type(8)))  float  v8f_t;

// Model dims
constexpr int Bc  = 2;
constexpr int Sq  = 512;
constexpr int Ee  = 512;
constexpr int Hh  = 8;
constexpr int Dd  = 64;
constexpr int FF  = 2048;
constexpr int MD  = 256;
constexpr int MS  = 65536;
constexpr int TK  = 32;
constexpr int Ntok = Bc * Sq;       // 1024
constexpr int L2v  = Sq * TK;       // 16384 retrieved KV per batch
constexpr int NKV  = Bc * L2v;      // 32768

// ---------------------------------------------------------------------------
// WMMA fragment helpers (CDNA5 16x16x32 bf16 layouts, wave32)
// A (16x32, MxK): lane L holds row L%16; elems j<8 -> K = 8*(L/16)+j,
//                 j>=8 -> K = 16 + 8*(L/16) + (j-8).
// B (32x16, KxN) loaded from an N-major (row n, K contiguous) LDS tile:
//                 lane L holds col L%16; same K order as A.
// C (16x16 f32): vgpr v, lane L -> (row = v + 8*(L/16), col = L%16).
// ---------------------------------------------------------------------------
__device__ inline v16bf_t load_a_frag(const bf16* __restrict__ base, int row_stride,
                                      int row0, int kbase, int lane) {
  v16bf_t a;
  const bf16* p = base + (long)(row0 + (lane & 15)) * row_stride + kbase + 8 * (lane >> 4);
#pragma unroll
  for (int j = 0; j < 8; ++j) a[j] = p[j];
#pragma unroll
  for (int j = 0; j < 8; ++j) a[8 + j] = p[16 + j];
  return a;
}

__device__ inline v16bf_t load_b_frag(const bf16* __restrict__ base, int row_stride,
                                      int col0, int kbase, int lane) {
  v16bf_t b;
  const bf16* p = base + (long)(col0 + (lane & 15)) * row_stride + kbase + 8 * (lane >> 4);
#pragma unroll
  for (int j = 0; j < 8; ++j) b[j] = p[j];
#pragma unroll
  for (int j = 0; j < 8; ++j) b[8 + j] = p[16 + j];
  return b;
}

// ---------------------------------------------------------------------------
// Generic GEMM: C[MxN] = A[MxK] * W[NxK]^T + bias, A fp32 or bf16, W bf16.
// BM=BN=64, BK=32; 256 threads = 8 waves; each wave owns two 16x16 C tiles.
// M,N multiples of 64; K multiple of 32 (true for every call site).
// ---------------------------------------------------------------------------
template <typename AT>
__global__ __launch_bounds__(256) void gemm_wmma(const AT* __restrict__ A,
                                                 const bf16* __restrict__ W,
                                                 const float* __restrict__ bias,
                                                 float* __restrict__ C,
                                                 int M, int N, int K) {
  __shared__ bf16 As[64 * 32];
  __shared__ bf16 Bs[64 * 32];
  const int tid = threadIdx.x, lane = tid & 31, wave = tid >> 5;
  const int m0 = blockIdx.y * 64, n0 = blockIdx.x * 64;
  const int tr = (wave >> 1) * 16;    // row tile
  const int tc = (wave & 1) * 32;     // col tiles tc, tc+16
  v8f_t acc0 = {}, acc1 = {};
  for (int k0 = 0; k0 < K; k0 += 32) {
    for (int i = tid; i < 64 * 32; i += 256) {
      int r = i >> 5, c = i & 31;
      As[i] = (bf16)(float)A[(long)(m0 + r) * K + k0 + c];
      Bs[i] = W[(long)(n0 + r) * K + k0 + c];
    }
    __syncthreads();
    v16bf_t a  = load_a_frag(As, 32, tr, 0, lane);
    v16bf_t b0 = load_b_frag(Bs, 32, tc, 0, lane);
    v16bf_t b1 = load_b_frag(Bs, 32, tc + 16, 0, lane);
    acc0 = __builtin_amdgcn_wmma_f32_16x16x32_bf16(false, a, false, b0, (short)0, acc0, false, false);
    acc1 = __builtin_amdgcn_wmma_f32_16x16x32_bf16(false, a, false, b1, (short)0, acc1, false, false);
    __syncthreads();
  }
  const int hf = lane >> 4, cl = lane & 15;
#pragma unroll
  for (int v = 0; v < 8; ++v) {
    int row = m0 + tr + v + 8 * hf;
    int c0  = n0 + tc + cl, c1 = c0 + 16;
    float o0 = acc0[v] + (bias ? bias[c0] : 0.f);
    float o1 = acc1[v] + (bias ? bias[c1] : 0.f);
    C[(long)row * N + c0] = o0;
    C[(long)row * N + c1] = o1;
  }
}

// ---------------------------------------------------------------------------
// Fused flash-style attention. Q:(B,H,S,D) K,V:(B,H,L,D) bf16, O:(B,S,H*D) f32.
// One WG = one (b,h) x 64-row q-tile, online softmax over L in chunks of 64.
// ---------------------------------------------------------------------------
__global__ __launch_bounds__(256) void attn_fused(const bf16* __restrict__ Q,
                                                  const bf16* __restrict__ Km,
                                                  const bf16* __restrict__ Vm,
                                                  float* __restrict__ O,
                                                  int S, int L, float scale) {
  __shared__ bf16 Qs[64 * 64];
  __shared__ bf16 Ks[64 * 64];
  __shared__ bf16 Vt[64 * 64];      // transposed: Vt[d*64 + kv]
  __shared__ bf16 Ps[64 * 64];
  __shared__ float Ss[64 * 64];
  __shared__ float mrow[64], lrow[64], arow[64];
  const int tid = threadIdx.x, lane = tid & 31, wave = tid >> 5;
  const int q0 = blockIdx.x * 64, h = blockIdx.y, b = blockIdx.z;
  const int H = gridDim.y, HD = H * 64;
  const long qbase  = (((long)b * H + h) * S + q0) * 64;
  const long kvbase = (((long)b * H + h) * (long)L) * 64;
  for (int i = tid; i < 64 * 64; i += 256) Qs[i] = Q[qbase + i];
  if (tid < 64) { mrow[tid] = -__builtin_inff(); lrow[tid] = 0.f; }
  const int tr = (wave >> 1) * 16, tc = (wave & 1) * 32;
  v8f_t o0 = {}, o1 = {};
  __syncthreads();

  for (int c0 = 0; c0 < L; c0 += 64) {
    for (int i = tid; i < 64 * 64; i += 256) {
      bf16 kk = Km[kvbase + (long)c0 * 64 + i];
      bf16 vv = Vm[kvbase + (long)c0 * 64 + i];
      Ks[i] = kk;
      int kv = i >> 6, d = i & 63;
      Vt[d * 64 + kv] = vv;
    }
    __syncthreads();
    // scores S = scale * Q K^T  (two 16x32 k-steps over D=64)
    v8f_t s0 = {}, s1 = {};
#pragma unroll
    for (int kk = 0; kk < 64; kk += 32) {
      v16bf_t a  = load_a_frag(Qs, 64, tr, kk, lane);
      v16bf_t b0 = load_b_frag(Ks, 64, tc, kk, lane);
      v16bf_t b1 = load_b_frag(Ks, 64, tc + 16, kk, lane);
      s0 = __builtin_amdgcn_wmma_f32_16x16x32_bf16(false, a, false, b0, (short)0, s0, false, false);
      s1 = __builtin_amdgcn_wmma_f32_16x16x32_bf16(false, a, false, b1, (short)0, s1, false, false);
    }
    {
      const int hf = lane >> 4, cl = lane & 15;
#pragma unroll
      for (int v = 0; v < 8; ++v) {
        Ss[(tr + v + 8 * hf) * 64 + tc + cl]      = s0[v] * scale;
        Ss[(tr + v + 8 * hf) * 64 + tc + 16 + cl] = s1[v] * scale;
      }
    }
    __syncthreads();
    // online softmax bookkeeping (one thread per row)
    if (tid < 64) {
      int r = tid;
      float mold = mrow[r], mc = mold;
      for (int j = 0; j < 64; ++j) mc = fmaxf(mc, Ss[r * 64 + j]);
      float alpha = (mold == -__builtin_inff()) ? 0.f : expf(mold - mc);
      float sum = 0.f;
      for (int j = 0; j < 64; ++j) {
        float p = expf(Ss[r * 64 + j] - mc);
        sum += p;
        Ps[r * 64 + j] = (bf16)p;
      }
      lrow[r] = lrow[r] * alpha + sum;
      mrow[r] = mc;
      arow[r] = alpha;
    }
    __syncthreads();
    // rescale accumulator, add P*V
    {
      const int hf = lane >> 4;
#pragma unroll
      for (int v = 0; v < 8; ++v) {
        float al = arow[tr + v + 8 * hf];
        o0[v] *= al; o1[v] *= al;
      }
    }
#pragma unroll
    for (int kk = 0; kk < 64; kk += 32) {
      v16bf_t a  = load_a_frag(Ps, 64, tr, kk, lane);
      v16bf_t b0 = load_b_frag(Vt, 64, tc, kk, lane);
      v16bf_t b1 = load_b_frag(Vt, 64, tc + 16, kk, lane);
      o0 = __builtin_amdgcn_wmma_f32_16x16x32_bf16(false, a, false, b0, (short)0, o0, false, false);
      o1 = __builtin_amdgcn_wmma_f32_16x16x32_bf16(false, a, false, b1, (short)0, o1, false, false);
    }
    __syncthreads();
  }
  const int hf = lane >> 4, cl = lane & 15;
#pragma unroll
  for (int v = 0; v < 8; ++v) {
    int r = tr + v + 8 * hf;
    float inv = 1.f / lrow[r];
    long orow = (long)(b * S + q0 + r) * HD + h * 64;
    O[orow + tc + cl]      = o0[v] * inv;
    O[orow + tc + 16 + cl] = o1[v] * inv;
  }
}

// ---------------------------------------------------------------------------
// Fused cosine-sim GEMM + running top-32. One WG = 32 query rows, streams the
// whole 64MB bf16 key bank (L2-resident across the 32 WGs).
// ---------------------------------------------------------------------------
__global__ __launch_bounds__(256) void sims_topk(const bf16* __restrict__ QN,
                                                 const bf16* __restrict__ KN,
                                                 int* __restrict__ outIdx) {
  __shared__ bf16 Qs[32 * 512];     // 32 KB
  __shared__ bf16 Bs[64 * 32];      //  4 KB key sub-tile
  __shared__ float Ss[32 * 64];     //  8 KB scores
  __shared__ float tval[32 * 32];
  __shared__ int   tidx[32 * 32];
  __shared__ float tmin[32];
  const int tid = threadIdx.x, lane = tid & 31, wave = tid >> 5;
  const int r0 = blockIdx.x * 32;
  for (int i = tid; i < 32 * 512; i += 256) Qs[i] = QN[(long)r0 * 512 + i];
  if (tid < 32) tmin[tid] = -__builtin_inff();
  for (int i = tid; i < 32 * 32; i += 256) { tval[i] = -__builtin_inff(); tidx[i] = 0; }
  const int tr = (wave >> 2) * 16;      // 2 row tiles
  const int tc = (wave & 3) * 16;       // 4 col tiles
  __syncthreads();

  for (long key0 = 0; key0 < MS; key0 += 64) {
    v8f_t acc = {};
    for (int kk = 0; kk < 512; kk += 32) {
      for (int i = tid; i < 64 * 32; i += 256) {
        int r = i >> 5, c = i & 31;
        Bs[i] = KN[(key0 + r) * 512 + kk + c];
      }
      __syncthreads();
      v16bf_t a = load_a_frag(Qs, 512, tr, kk, lane);
      v16bf_t b = load_b_frag(Bs, 32, tc, 0, lane);
      acc = __builtin_amdgcn_wmma_f32_16x16x32_bf16(false, a, false, b, (short)0, acc, false, false);
      __syncthreads();
    }
    {
      const int hf = lane >> 4, cl = lane & 15;
#pragma unroll
      for (int v = 0; v < 8; ++v)
        Ss[(tr + v + 8 * hf) * 64 + tc + cl] = acc[v];
    }
    __syncthreads();
    if (tid < 32) {
      int r = tid;
      float mn = tmin[r];
      for (int j = 0; j < 64; ++j) {
        float s = Ss[r * 64 + j];
        if (s > mn) {
          int slot = 0; float mv = tval[r * 32];
          for (int q = 1; q < 32; ++q) {
            float v2 = tval[r * 32 + q];
            if (v2 < mv) { mv = v2; slot = q; }
          }
          tval[r * 32 + slot] = s;
          tidx[r * 32 + slot] = (int)key0 + j;
          mv = tval[r * 32];
          for (int q = 1; q < 32; ++q) mv = fminf(mv, tval[r * 32 + q]);
          mn = mv;
        }
      }
      tmin[r] = mn;
    }
    __syncthreads();
  }
  // attention over the retrieved set is permutation-invariant -> no sort needed
  for (int i = tid; i < 32 * 32; i += 256) outIdx[(long)r0 * TK + i] = tidx[i];
}

// ---------------------------------------------------------------------------
// Support kernels
// ---------------------------------------------------------------------------
__global__ void to_bf16_kernel(const float* __restrict__ in, bf16* __restrict__ out, long n) {
  long i = (long)blockIdx.x * 256 + threadIdx.x;
  if (i < n) out[i] = (bf16)in[i];
}

__global__ void transpose_bf16_kernel(const bf16* __restrict__ in, bf16* __restrict__ out,
                                      int rows, int cols) {
  long i = (long)blockIdx.x * 256 + threadIdx.x;
  if (i >= (long)rows * cols) return;
  int r = (int)(i / cols), c = (int)(i % cols);
  out[(long)c * rows + r] = in[i];
}

__global__ void split_heads_kernel(const float* __restrict__ in, bf16* __restrict__ out,
                                   int Sx, long total) {
  long i = (long)blockIdx.x * 256 + threadIdx.x;
  if (i >= total) return;
  int d = (int)(i & 63);
  int h = (int)((i >> 6) & 7);
  long rest = i >> 9;               // b*Sx + s
  long s = rest % Sx, b = rest / Sx;
  out[(((b * Hh + h) * (long)Sx + s) << 6) + d] = (bf16)in[i];
}

__global__ void add_scaled_kernel(float* __restrict__ out, const float* __restrict__ a,
                                  const float* __restrict__ b, float beta, long n) {
  long i = (long)blockIdx.x * 256 + threadIdx.x;
  if (i < n) out[i] = a[i] + beta * b[i];
}

__global__ void gelu_kernel(float* __restrict__ x, long n) {
  long i = (long)blockIdx.x * 256 + threadIdx.x;
  if (i < n) {
    float v = x[i];
    x[i] = 0.5f * v * (1.0f + erff(v * 0.70710678118654752f));  // exact erf GELU
  }
}

__global__ __launch_bounds__(256) void softmax_rows(float* __restrict__ x, int cols, float scale) {
  __shared__ float red[256];
  long base = (long)blockIdx.x * cols;
  float m = -__builtin_inff();
  for (int j = threadIdx.x; j < cols; j += 256) m = fmaxf(m, x[base + j] * scale);
  red[threadIdx.x] = m; __syncthreads();
  for (int s = 128; s > 0; s >>= 1) {
    if (threadIdx.x < s) red[threadIdx.x] = fmaxf(red[threadIdx.x], red[threadIdx.x + s]);
    __syncthreads();
  }
  m = red[0]; __syncthreads();
  float sum = 0.f;
  for (int j = threadIdx.x; j < cols; j += 256) {
    float e = expf(x[base + j] * scale - m);
    x[base + j] = e; sum += e;
  }
  red[threadIdx.x] = sum; __syncthreads();
  for (int s = 128; s > 0; s >>= 1) {
    if (threadIdx.x < s) red[threadIdx.x] += red[threadIdx.x + s];
    __syncthreads();
  }
  float inv = 1.f / red[0];
  for (int j = threadIdx.x; j < cols; j += 256) x[base + j] *= inv;
}

__global__ __launch_bounds__(256) void layernorm_res(const float* __restrict__ a,
                                                     const float* __restrict__ r,
                                                     const float* __restrict__ g,
                                                     const float* __restrict__ bb,
                                                     float* __restrict__ out, int cols) {
  __shared__ float red[256];
  __shared__ float mu_s, rstd_s;
  long base = (long)blockIdx.x * cols;
  float s = 0.f;
  for (int j = threadIdx.x; j < cols; j += 256) s += a[base + j] + r[base + j];
  red[threadIdx.x] = s; __syncthreads();
  for (int st = 128; st > 0; st >>= 1) {
    if (threadIdx.x < st) red[threadIdx.x] += red[threadIdx.x + st];
    __syncthreads();
  }
  if (threadIdx.x == 0) mu_s = red[0] / cols;
  __syncthreads();
  float mu = mu_s, v = 0.f;
  for (int j = threadIdx.x; j < cols; j += 256) {
    float d = a[base + j] + r[base + j] - mu;
    v += d * d;
  }
  red[threadIdx.x] = v; __syncthreads();
  for (int st = 128; st > 0; st >>= 1) {
    if (threadIdx.x < st) red[threadIdx.x] += red[threadIdx.x + st];
    __syncthreads();
  }
  if (threadIdx.x == 0) rstd_s = rsqrtf(red[0] / cols + 1e-5f);
  __syncthreads();
  float rstd = rstd_s;
  for (int j = threadIdx.x; j < cols; j += 256)
    out[base + j] = (a[base + j] + r[base + j] - mu) * rstd * g[j] + bb[j];
}

__global__ __launch_bounds__(256) void l2norm_rows(const float* __restrict__ in,
                                                   bf16* __restrict__ out, int cols) {
  __shared__ float red[256];
  long base = (long)blockIdx.x * cols;
  float s = 0.f;
  for (int j = threadIdx.x; j < cols; j += 256) { float v = in[base + j]; s += v * v; }
  red[threadIdx.x] = s; __syncthreads();
  for (int st = 128; st > 0; st >>= 1) {
    if (threadIdx.x < st) red[threadIdx.x] += red[threadIdx.x + st];
    __syncthreads();
  }
  float inv = 1.f / fmaxf(sqrtf(red[0]), 1e-12f);
  for (int j = threadIdx.x; j < cols; j += 256) out[base + j] = (bf16)(in[base + j] * inv);
}

__global__ void gather_rows_kernel(const float* __restrict__ vals, const int* __restrict__ idx,
                                   bf16* __restrict__ out, long total) {
  long i = (long)blockIdx.x * 256 + threadIdx.x;
  if (i >= total) return;
  int e = (int)(i & 511);
  long row = i >> 9;
  long id = idx[row];
  out[i] = (bf16)vals[id * 512 + e];
}

// ---------------------------------------------------------------------------
// Orchestration
// ---------------------------------------------------------------------------
enum {
  IN_X = 0,
  IA_QW, IA_QB, IA_KW, IA_KB, IA_VW, IA_VB, IA_OW, IA_OB,
  SA_QW, SA_QB, SA_KW, SA_KB, SA_VW, SA_VB, SA_OW, SA_OB,
  MA_QW, MA_QB, MA_KW, MA_KB, MA_VW, MA_VB, MA_OW, MA_OB,
  IT_FC1W, IT_FC1B, IT_FC2W, IT_FC2B, MT_FC1W, MT_FC1B, MT_FC2W, MT_FC2B,
  LN1_G, LN1_B, LN2_G, LN2_B, MN1_G, MN1_B, MN2_G, MN2_B, MN3_G, MN3_B,
  MEM_STATE, MEM_KEYS, MEM_VALUES
};

static inline void* bump(char** p, size_t bytes) {
  void* r = *p;
  *p += (bytes + 255) & ~(size_t)255;
  return r;
}

extern "C" void kernel_launch(void* const* d_in, const int* in_sizes, int n_in,
                              void* d_out, int out_size, void* d_ws, size_t ws_size,
                              hipStream_t stream) {
  (void)in_sizes; (void)n_in; (void)out_size; (void)ws_size;
  const float scale = 0.125f;  // D^-0.5

  char* p = (char*)d_ws;
  // bf16 weights
  bf16* Wiq = (bf16*)bump(&p, (size_t)Ee * Ee * 2);
  bf16* Wik = (bf16*)bump(&p, (size_t)Ee * Ee * 2);
  bf16* Wiv = (bf16*)bump(&p, (size_t)Ee * Ee * 2);
  bf16* Wio = (bf16*)bump(&p, (size_t)Ee * Ee * 2);
  bf16* Wsq = (bf16*)bump(&p, (size_t)Ee * Ee * 2);
  bf16* Wsk = (bf16*)bump(&p, (size_t)Ee * Ee * 2);
  bf16* Wsv = (bf16*)bump(&p, (size_t)Ee * Ee * 2);
  bf16* Wso = (bf16*)bump(&p, (size_t)Ee * Ee * 2);
  bf16* Wmq = (bf16*)bump(&p, (size_t)Ee * Ee * 2);
  bf16* Wmk = (bf16*)bump(&p, (size_t)Ee * Ee * 2);
  bf16* Wmv = (bf16*)bump(&p, (size_t)Ee * Ee * 2);
  bf16* Wmo = (bf16*)bump(&p, (size_t)Ee * Ee * 2);
  bf16* Wf1 = (bf16*)bump(&p, (size_t)FF * Ee * 2);
  bf16* Wf2 = (bf16*)bump(&p, (size_t)Ee * FF * 2);
  bf16* Wg1 = (bf16*)bump(&p, (size_t)FF * Ee * 2);
  bf16* Wg2 = (bf16*)bump(&p, (size_t)Ee * FF * 2);
  bf16* Wms  = (bf16*)bump(&p, (size_t)MD * Ee * 2);
  bf16* WmsT = (bf16*)bump(&p, (size_t)Ee * MD * 2);
  bf16* KNbf = (bf16*)bump(&p, (size_t)MS * Ee * 2);   // 64 MB, L2-resident
  bf16* QNbf = (bf16*)bump(&p, (size_t)Ntok * Ee * 2);
  // f32 activations
  float* qf    = (float*)bump(&p, (size_t)Ntok * Ee * 4);
  float* kf    = (float*)bump(&p, (size_t)Ntok * Ee * 4);
  float* vf    = (float*)bump(&p, (size_t)Ntok * Ee * 4);
  float* attnO = (float*)bump(&p, (size_t)Ntok * Ee * 4);
  float* memO  = (float*)bump(&p, (size_t)Ntok * Ee * 4);
  float* tmp1  = (float*)bump(&p, (size_t)Ntok * Ee * 4);
  float* tA    = (float*)bump(&p, (size_t)Ntok * Ee * 4);
  float* memw  = (float*)bump(&p, (size_t)Ntok * MD * 4);
  float* ffn   = (float*)bump(&p, (size_t)Ntok * FF * 4);
  float* bufA  = (float*)bump(&p, (size_t)Ntok * Ee * 4);
  float* bufB  = (float*)bump(&p, (size_t)Ntok * Ee * 4);
  float* bufC  = (float*)bump(&p, (size_t)Ntok * Ee * 4);
  float* bufD  = (float*)bump(&p, (size_t)Ntok * Ee * 4);
  // head-split bf16
  bf16* qh  = (bf16*)bump(&p, (size_t)Ntok * Ee * 2);
  bf16* kh  = (bf16*)bump(&p, (size_t)Ntok * Ee * 2);
  bf16* vh  = (bf16*)bump(&p, (size_t)Ntok * Ee * 2);
  bf16* k2h = (bf16*)bump(&p, (size_t)NKV * Ee * 2);   // 32 MB
  bf16* v2h = (bf16*)bump(&p, (size_t)NKV * Ee * 2);   // 32 MB
  bf16* rflat = (bf16*)bump(&p, (size_t)NKV * Ee * 2); // 32 MB
  float* bigf = (float*)bump(&p, (size_t)NKV * Ee * 4);// 64 MB (reused for k2/v2)
  int* topIdx = (int*)bump(&p, (size_t)Ntok * TK * 4);

  auto cvt = [&](int idx, bf16* dst, long n) {
    to_bf16_kernel<<<(int)((n + 255) / 256), 256, 0, stream>>>((const float*)d_in[idx], dst, n);
  };
  auto gemmF = [&](const float* A, const bf16* W, const float* bias, float* C,
                   int M, int N, int K) {
    dim3 g(N / 64, M / 64, 1);
    gemm_wmma<float><<<g, 256, 0, stream>>>(A, W, bias, C, M, N, K);
  };
  auto gemmH = [&](const bf16* A, const bf16* W, const float* bias, float* C,
                   int M, int N, int K) {
    dim3 g(N / 64, M / 64, 1);
    gemm_wmma<bf16><<<g, 256, 0, stream>>>(A, W, bias, C, M, N, K);
  };
  auto split = [&](const float* in, bf16* out, int Sx) {
    long total = (long)Bc * Sx * Ee;
    split_heads_kernel<<<(int)((total + 255) / 256), 256, 0, stream>>>(in, out, Sx, total);
  };
  auto lnres = [&](const float* a, const float* r, int gi, int bi, float* out) {
    layernorm_res<<<Ntok, 256, 0, stream>>>(a, r, (const float*)d_in[gi],
                                            (const float*)d_in[bi], out, Ee);
  };
  const float* x = (const float*)d_in[IN_X];

  // --- weight conversion (deterministic, every launch) ---
  cvt(IA_QW, Wiq, (long)Ee * Ee); cvt(IA_KW, Wik, (long)Ee * Ee);
  cvt(IA_VW, Wiv, (long)Ee * Ee); cvt(IA_OW, Wio, (long)Ee * Ee);
  cvt(SA_QW, Wsq, (long)Ee * Ee); cvt(SA_KW, Wsk, (long)Ee * Ee);
  cvt(SA_VW, Wsv, (long)Ee * Ee); cvt(SA_OW, Wso, (long)Ee * Ee);
  cvt(MA_QW, Wmq, (long)Ee * Ee); cvt(MA_KW, Wmk, (long)Ee * Ee);
  cvt(MA_VW, Wmv, (long)Ee * Ee); cvt(MA_OW, Wmo, (long)Ee * Ee);
  cvt(IT_FC1W, Wf1, (long)FF * Ee); cvt(IT_FC2W, Wf2, (long)Ee * FF);
  cvt(MT_FC1W, Wg1, (long)FF * Ee); cvt(MT_FC2W, Wg2, (long)Ee * FF);
  cvt(MEM_STATE, Wms, (long)MD * Ee);
  transpose_bf16_kernel<<<(int)(((long)MD * Ee + 255) / 256), 256, 0, stream>>>(Wms, WmsT, MD, Ee);
  l2norm_rows<<<MS, 256, 0, stream>>>((const float*)d_in[MEM_KEYS], KNbf, Ee);

  // --- InfiniTransformer layer ---
  gemmF(x, Wiq, (const float*)d_in[IA_QB], qf, Ntok, Ee, Ee);
  gemmF(x, Wik, (const float*)d_in[IA_KB], kf, Ntok, Ee, Ee);
  gemmF(x, Wiv, (const float*)d_in[IA_VB], vf, Ntok, Ee, Ee);
  split(qf, qh, Sq); split(kf, kh, Sq); split(vf, vh, Sq);
  { dim3 g(Sq / 64, Hh, Bc);
    attn_fused<<<g, 256, 0, stream>>>(qh, kh, vh, attnO, Sq, Sq, scale); }
  gemmF(x, Wms, nullptr, memw, Ntok, MD, Ee);
  softmax_rows<<<Ntok, 256, 0, stream>>>(memw, MD, scale);
  gemmF(memw, WmsT, nullptr, memO, Ntok, Ee, MD);
  add_scaled_kernel<<<(int)(((long)Ntok * Ee + 255) / 256), 256, 0, stream>>>(
      tmp1, attnO, memO, 0.5f, (long)Ntok * Ee);
  gemmF(tmp1, Wio, (const float*)d_in[IA_OB], tA, Ntok, Ee, Ee);
  lnres(x, tA, LN1_G, LN1_B, bufA);
  gemmF(bufA, Wf1, (const float*)d_in[IT_FC1B], ffn, Ntok, FF, Ee);
  gelu_kernel<<<(int)(((long)Ntok * FF + 255) / 256), 256, 0, stream>>>(ffn, (long)Ntok * FF);
  gemmF(ffn, Wf2, (const float*)d_in[IT_FC2B], tA, Ntok, Ee, FF);
  lnres(bufA, tA, LN2_G, LN2_B, bufB);

  // --- Memorizing layer: self-attention ---
  gemmF(bufB, Wsq, (const float*)d_in[SA_QB], qf, Ntok, Ee, Ee);
  gemmF(bufB, Wsk, (const float*)d_in[SA_KB], kf, Ntok, Ee, Ee);
  gemmF(bufB, Wsv, (const float*)d_in[SA_VB], vf, Ntok, Ee, Ee);
  split(qf, qh, Sq); split(kf, kh, Sq); split(vf, vh, Sq);
  { dim3 g(Sq / 64, Hh, Bc);
    attn_fused<<<g, 256, 0, stream>>>(qh, kh, vh, attnO, Sq, Sq, scale); }
  gemmF(attnO, Wso, (const float*)d_in[SA_OB], tA, Ntok, Ee, Ee);
  lnres(bufB, tA, MN1_G, MN1_B, bufC);

  // --- kNN retrieval: fused sims + top-32, then gather ---
  l2norm_rows<<<Ntok, 256, 0, stream>>>(bufC, QNbf, Ee);
  sims_topk<<<Ntok / 32, 256, 0, stream>>>(QNbf, KNbf, topIdx);
  { long total = (long)NKV * Ee;
    gather_rows_kernel<<<(int)((total + 255) / 256), 256, 0, stream>>>(
        (const float*)d_in[MEM_VALUES], topIdx, rflat, total); }

  // --- memory cross-attention over retrieved bank (L = 16384 per batch) ---
  gemmF(bufC, Wmq, (const float*)d_in[MA_QB], qf, Ntok, Ee, Ee);
  split(qf, qh, Sq);
  gemmH(rflat, Wmk, (const float*)d_in[MA_KB], bigf, NKV, Ee, Ee);
  split(bigf, k2h, L2v);
  gemmH(rflat, Wmv, (const float*)d_in[MA_VB], bigf, NKV, Ee, Ee);
  split(bigf, v2h, L2v);
  { dim3 g(Sq / 64, Hh, Bc);
    attn_fused<<<g, 256, 0, stream>>>(qh, k2h, v2h, attnO, Sq, L2v, scale); }
  gemmF(attnO, Wmo, (const float*)d_in[MA_OB], tA, Ntok, Ee, Ee);
  lnres(bufC, tA, MN2_G, MN2_B, bufD);

  // --- final FFN + LN -> d_out ---
  gemmF(bufD, Wg1, (const float*)d_in[MT_FC1B], ffn, Ntok, FF, Ee);
  gelu_kernel<<<(int)(((long)Ntok * FF + 255) / 256), 256, 0, stream>>>(ffn, (long)Ntok * FF);
  gemmF(ffn, Wg2, (const float*)d_in[MT_FC2B], tA, Ntok, Ee, FF);
  lnres(bufD, tA, MN3_G, MN3_B, (float*)d_out);
}
